// GRBM_25288767439222
// MI455X (gfx1250) — matrix operand order repaired and checked by reading, not compiled, not run on hardware
//
#include <hip/hip_runtime.h>
#include <hip/hip_bf16.h>

// ---------------------------------------------------------------------------
// GRBM Gibbs sampling on gfx1250 (MI455X): bf16 WMMA GEMM chain.
//   B=4096 (batch), V=3072 (visible), H=512 (hidden), 8 CD steps.
//   Async global->LDS (ASYNCcnt) double-buffered pipeline; the gfx1250
//   builtin takes (AS1 v4i* src, AS3 v4i* dst, imm offset, imm cpol).
// ---------------------------------------------------------------------------

#ifndef __has_builtin
#define __has_builtin(x) 0
#endif

#if defined(__AMDGCN__) && __has_builtin(__builtin_amdgcn_global_load_async_to_lds_b128)
#define USE_ASYNC_LDS 1
#else
#define USE_ASYNC_LDS 0
#endif

typedef __attribute__((ext_vector_type(16))) __bf16   v16bf;
typedef __attribute__((ext_vector_type(8)))  float    v8f;
typedef __attribute__((ext_vector_type(4)))  unsigned u32x4;

// exact GCC-vector type the async builtin expects (per clang diagnostic)
typedef int gv4i __attribute__((vector_size(16)));
#define ASYNC_GPTR(p) ((__attribute__((address_space(1))) gv4i*)(p))
#define ASYNC_LPTR(p) ((__attribute__((address_space(3))) gv4i*)(p))

union FragBF {
    v16bf v;
    u32x4 q[2];
};

static constexpr int kB = 4096;
static constexpr int kV = 3072;
static constexpr int kH = 512;

static constexpr int BM = 128;
static constexpr int BN = 128;
static constexpr int BK = 32;
static constexpr int LDSS = BK + 8;   // 40 elems = 80B rows -> 16B aligned, bank-staggered

// --- counter-based RNG helpers ---------------------------------------------
__device__ __forceinline__ unsigned hash32(unsigned x) {
    x ^= x >> 16; x *= 0x7feb352du;
    x ^= x >> 15; x *= 0x846ca68bu;
    x ^= x >> 16;
    return x;
}
__device__ __forceinline__ float u01(unsigned h) {          // [0,1)
    return (float)(h >> 8) * (1.0f / 16777216.0f);
}
__device__ __forceinline__ float u01p(unsigned h) {         // (0,1]
    return (float)((h >> 8) + 1u) * (1.0f / 16777217.0f);
}

__device__ __forceinline__ void wait_async_zero() {
#if defined(__AMDGCN__)
#if __has_builtin(__builtin_amdgcn_s_wait_asynccnt)
    __builtin_amdgcn_s_wait_asynccnt(0);
#else
    asm volatile("s_wait_asynccnt 0x0" ::: "memory");
#endif
#endif
}

// --- one K-slab of WMMA work (frag loads per ISA 7.12.2 + 8 WMMAs) ----------
__device__ __forceinline__ void compute_slab(const __bf16* __restrict__ As,
                                             const __bf16* __restrict__ Bs,
                                             int wm0, int wn0, int n_lane, int half,
                                             v8f (&acc)[2][4])
{
    FragBF a[2], b[4];
#pragma unroll
    for (int mt = 0; mt < 2; ++mt) {
        const __bf16* p = As + (wm0 + mt * 16 + n_lane) * LDSS;
        a[mt].q[0] = *(const u32x4*)(p + half * 8);        // K = half*8 + 0..7
        a[mt].q[1] = *(const u32x4*)(p + 16 + half * 8);   // K = 16 + half*8 + 0..7
    }
#pragma unroll
    for (int nt = 0; nt < 4; ++nt) {
        const __bf16* p = Bs + (wn0 + nt * 16 + n_lane) * LDSS;
        b[nt].q[0] = *(const u32x4*)(p + half * 8);        // K = half*8 + 0..7
        b[nt].q[1] = *(const u32x4*)(p + 16 + half * 8);   // K = 16 + half*8 + 0..7
    }
#pragma unroll
    for (int mt = 0; mt < 2; ++mt)
#pragma unroll
        for (int nt = 0; nt < 4; ++nt)
            acc[mt][nt] = __builtin_amdgcn_wmma_f32_16x16x32_bf16(
                false, a[mt].v, false, b[nt].v,
                (short)0, acc[mt][nt], false, false);
}

// ---------------------------------------------------------------------------
// Generic tiled WMMA GEMM:  C[M,N] = A[M,K] * Bt[N,K]^T   (both row-major, ld=K)
// EPI=0 : h-update  ->  outBF[m*N+n] = bernoulli(sigmoid(acc + bias[n]))
// EPI=1 : v-update  ->  val = acc + mu[n] + gauss*std[n];
//                       outF[m*N+n] = val; outBF[m*N+n] = bf16(val/var[n])
// ---------------------------------------------------------------------------
template <int EPI>
__global__ __launch_bounds__(256)
void gemm_wmma(const __bf16* __restrict__ A,
               const __bf16* __restrict__ Bt,
               int M, int N, int K,
               const float* __restrict__ bias,
               const float* __restrict__ mu,
               const float* __restrict__ logvar,
               float*       __restrict__ outF,
               __bf16*      __restrict__ outBF,
               unsigned seed)
{
    const int tid    = threadIdx.x;
    const int lane   = tid & 31;
    const int wave   = tid >> 5;
    const int n_lane = lane & 15;   // column / M-row-in-tile selector
    const int half   = lane >> 4;   // lane half (K / M split per ISA layout)

    const int bn0 = blockIdx.x * BN;
    const int bm0 = blockIdx.y * BM;

    const int wm0 = (wave & 3) * 32;  // 4 waves along M (2 m-tiles each)
    const int wn0 = (wave >> 2) * 64; // 2 waves along N (4 n-tiles each)

    const int kIters = K / BK;

    // global<->LDS staging map: 128 rows x 32 bf16 = 512 chunks of 16B; 2/thread
    int grow[2], gcol[2];
#pragma unroll
    for (int i = 0; i < 2; ++i) {
        int c   = tid + i * 256;
        grow[i] = c >> 2;
        gcol[i] = (c & 3) * 8;
    }

    v8f acc[2][4] = {};

#if USE_ASYNC_LDS
    // ---- ASYNCcnt double-buffered pipeline: global -> LDS directly ----
    __shared__ __bf16 As[2][BM * LDSS];
    __shared__ __bf16 Bs[2][BN * LDSS];

#define ISSUE_SLAB(KT, BUF)                                                          \
    do {                                                                             \
        const int _ko = (KT) * BK;                                                   \
        _Pragma("unroll")                                                            \
        for (int i = 0; i < 2; ++i) {                                                \
            __builtin_amdgcn_global_load_async_to_lds_b128(                          \
                ASYNC_GPTR(A + (size_t)(bm0 + grow[i]) * K + _ko + gcol[i]),         \
                ASYNC_LPTR(&As[(BUF)][grow[i] * LDSS + gcol[i]]), 0, 0);             \
            __builtin_amdgcn_global_load_async_to_lds_b128(                          \
                ASYNC_GPTR(Bt + (size_t)(bn0 + grow[i]) * K + _ko + gcol[i]),        \
                ASYNC_LPTR(&Bs[(BUF)][grow[i] * LDSS + gcol[i]]), 0, 0);             \
        }                                                                            \
    } while (0)

    ISSUE_SLAB(0, 0);
    wait_async_zero();
    __syncthreads();

    for (int kt = 0; kt < kIters; ++kt) {
        if (kt + 1 < kIters)
            ISSUE_SLAB(kt + 1, (kt + 1) & 1);         // fill other buffer
        compute_slab(As[kt & 1], Bs[kt & 1], wm0, wn0, n_lane, half, acc);
        if (kt + 1 < kIters)
            wait_async_zero();                        // my slab-(kt+1) copies done
        __syncthreads();                              // everyone's copies visible
    }
#undef ISSUE_SLAB
#else
    // ---- fallback: register staging, single LDS buffer ----
    __shared__ __bf16 As[1][BM * LDSS];
    __shared__ __bf16 Bs[1][BN * LDSS];

    u32x4 regA[2], regB[2];
#pragma unroll
    for (int i = 0; i < 2; ++i) {
        regA[i] = *(const u32x4*)(A  + (size_t)(bm0 + grow[i]) * K + gcol[i]);
        regB[i] = *(const u32x4*)(Bt + (size_t)(bn0 + grow[i]) * K + gcol[i]);
    }

    for (int kt = 0; kt < kIters; ++kt) {
#pragma unroll
        for (int i = 0; i < 2; ++i) {
            *(u32x4*)(&As[0][grow[i] * LDSS + gcol[i]]) = regA[i];
            *(u32x4*)(&Bs[0][grow[i] * LDSS + gcol[i]]) = regB[i];
        }
        __syncthreads();

        if (kt + 1 < kIters) {
            const int ko = (kt + 1) * BK;
#pragma unroll
            for (int i = 0; i < 2; ++i) {
                regA[i] = *(const u32x4*)(A  + (size_t)(bm0 + grow[i]) * K + ko + gcol[i]);
                regB[i] = *(const u32x4*)(Bt + (size_t)(bn0 + grow[i]) * K + ko + gcol[i]);
            }
        }
        if (kt + 2 < kIters) {
            const int ko2 = (kt + 2) * BK;
            __builtin_prefetch(A  + (size_t)(bm0 + (tid >> 1)) * K + ko2, 0, 1);
            __builtin_prefetch(Bt + (size_t)(bn0 + (tid >> 1)) * K + ko2, 0, 1);
        }

        compute_slab(As[0], Bs[0], wm0, wn0, n_lane, half, acc);
        __syncthreads();
    }
#endif

    // ---- fused sampling epilogue ----
#pragma unroll
    for (int mt = 0; mt < 2; ++mt) {
#pragma unroll
        for (int nt = 0; nt < 4; ++nt) {
            const int n = bn0 + wn0 + nt * 16 + n_lane;
#pragma unroll
            for (int r = 0; r < 8; ++r) {
                const int m = bm0 + wm0 + mt * 16 + half * 8 + r;
                float accv  = acc[mt][nt][r];
                const unsigned idx = (unsigned)m * (unsigned)N + (unsigned)n;
                if (EPI == 0) {
                    // p = sigmoid(acc + b[n]);  h = 1[u < p]
                    float p = 1.0f / (1.0f + __expf(-(accv + bias[n])));
                    float u = u01(hash32(seed ^ idx));
                    outBF[(size_t)m * N + n] = (__bf16)(u < p ? 1.0f : 0.0f);
                } else {
                    float var = fmaxf(__expf(logvar[n]), 1e-8f);
                    float sd  = sqrtf(var);
                    float f1  = u01p(hash32(seed ^ idx));
                    float f2  = u01(hash32((seed * 0x9E3779B9u) ^ idx ^ 0x85EBCA6Bu));
                    float g   = sqrtf(-2.0f * __logf(f1)) * __cosf(6.28318530718f * f2);
                    float val = accv + mu[n] + g * sd;
                    outF [(size_t)m * N + n] = val;                 // sample (f32 output)
                    outBF[(size_t)m * N + n] = (__bf16)(val / var); // scaled feed for next h-update
                }
            }
        }
    }
}

// --- one-time conversions ---------------------------------------------------
__global__ void cvt_w(const float* __restrict__ W,
                      __bf16* __restrict__ Wb, __bf16* __restrict__ WTb)
{
    int idx = blockIdx.x * blockDim.x + threadIdx.x;
    if (idx >= kV * kH) return;
    int r = idx / kH, c = idx % kH;
    float w = W[idx];
    Wb[idx]                 = (__bf16)w;   // [V][H]
    WTb[(size_t)c * kV + r] = (__bf16)w;   // [H][V]
}

__global__ void cvt_v(const float* __restrict__ v, const float* __restrict__ logvar,
                      __bf16* __restrict__ Vb)
{
    size_t idx = (size_t)blockIdx.x * blockDim.x + threadIdx.x;
    if (idx >= (size_t)kB * kV) return;
    int c = (int)(idx % kV);
    float var = fmaxf(__expf(logvar[c]), 1e-8f);
    Vb[idx] = (__bf16)(v[idx] / var);
}

// ---------------------------------------------------------------------------
extern "C" void kernel_launch(void* const* d_in, const int* in_sizes, int n_in,
                              void* d_out, int out_size, void* d_ws, size_t ws_size,
                              hipStream_t stream)
{
    (void)in_sizes; (void)n_in; (void)out_size; (void)ws_size;

    const float* v  = (const float*)d_in[0];   // [B,V]
    const float* W  = (const float*)d_in[1];   // [V,H]
    const float* b  = (const float*)d_in[2];   // [H]
    const float* mu = (const float*)d_in[3];   // [V]
    const float* lv = (const float*)d_in[4];   // [V]
    float* out      = (float*)d_out;           // [8,B,V]

    __bf16* Wb  = (__bf16*)d_ws;               // V*H
    __bf16* WTb = Wb  + (size_t)kV * kH;       // H*V
    __bf16* Vb  = WTb + (size_t)kV * kH;       // B*V (holds v/var in bf16)
    __bf16* Hb  = Vb  + (size_t)kB * kV;       // B*H

    cvt_w<<<(kV * kH + 255) / 256, 256, 0, stream>>>(W, Wb, WTb);
    cvt_v<<<(int)(((size_t)kB * kV + 255) / 256), 256, 0, stream>>>(v, lv, Vb);

    dim3 blk(256);
    dim3 gridH(kH / BN, kB / BM);   // h-update: M=B, N=H, K=V
    dim3 gridV(kV / BN, kB / BM);   // v-update: M=B, N=V, K=H

    // initial h0 = bernoulli(sigmoid((v/var) @ W + b))
    gemm_wmma<0><<<gridH, blk, 0, stream>>>(Vb, WTb, kB, kH, kV,
                                            b, nullptr, nullptr, nullptr, Hb,
                                            0xC0FFEE01u);
    for (int s = 0; s < 8; ++s) {
        // v_s = h @ W^T + mu + noise   (writes output slice s and Vb = v/var)
        gemm_wmma<1><<<gridV, blk, 0, stream>>>(Hb, Wb, kB, kV, kH,
                                                nullptr, mu, lv,
                                                out + (size_t)s * kB * kV, Vb,
                                                0x9E3779B9u * (unsigned)(s + 1));
        if (s < 7) {
            // h = bernoulli(sigmoid((v_s/var) @ W + b)); final h discarded by ref
            gemm_wmma<0><<<gridH, blk, 0, stream>>>(Vb, WTb, kB, kH, kV,
                                                    b, nullptr, nullptr, nullptr, Hb,
                                                    0x85EBCA6Bu * (unsigned)(s + 2));
        }
    }
}
